// MC_GRU_BN_predictor_78795470012743
// MI455X (gfx1250) — matrix operand order repaired and checked by reading, not compiled, not run on hardware
//
#include <hip/hip_runtime.h>
#include <hip/hip_bf16.h>

typedef __attribute__((ext_vector_type(16))) __bf16 v16bf;
typedef __attribute__((ext_vector_type(8)))  __bf16 v8bf;
typedef __attribute__((ext_vector_type(8)))  float  v8f;

#define B_   256
#define T_   512
#define C_   34
#define H_   64
#define G3   192      // 3*H
#define BM   16       // batch tile per workgroup
#define ACCW 200      // padded row stride (floats) for gate-preact LDS
#define HBW  72       // padded row stride (bf16 elems) for hidden-state LDS

// ---------------------------------------------------------------------------
// Kernel 1: per-(channel, batch-tile) GRU scan over T with WMMA recurrence.
// One workgroup = 128 threads = 4 wave32s; each wave owns 3 of 12 N-tiles.
// ---------------------------------------------------------------------------
__global__ __launch_bounds__(128)
void gru_scan_kernel(const float* __restrict__ x,
                     const float* __restrict__ Wih,
                     const float* __restrict__ Whh,
                     const float* __restrict__ b_ih,
                     const float* __restrict__ b_hh,
                     const int*   __restrict__ lens,
                     float*       __restrict__ hfinal)
{
    __shared__ float  accS[BM * ACCW];                       // gate pre-activations [16 x 192]
    __shared__ float  hf[BM * H_];                           // fp32 hidden state
    __shared__ __attribute__((aligned(16))) __bf16 hb[BM * HBW]; // bf16 hidden (A-frag source)
    __shared__ float  xs[BM];                                // x_t slice for the 16 rows

    const int tid  = threadIdx.x;
    const int lane = tid & 31;
    const int w    = tid >> 5;          // wave 0..3
    const int hi   = lane >> 4;         // half-wave select
    const int col  = lane & 15;         // N column within tile / M row for A

    const int c  = blockIdx.x % C_;
    const int b0 = (blockIdx.x / C_) * BM;

    // ---- one-time preloads ------------------------------------------------
    // B fragments: B[k,n] = Whh[c, g=n, h=k]; dense 32x16 bf16 layout:
// lane col = n (tile*16+col), elem e holds K = 8*hi + e (e<8) / 16 + 8*hi + (e-8) (e>=8) (+32 per k-chunk).
    v16bf Bf[3][2];
    float cmul[3], cbase[3];
#pragma unroll
    for (int s = 0; s < 3; ++s) {
        const int g = (w * 3 + s) * 16 + col;                // gate row 0..191
        const float* wrow = Whh + ((size_t)c * G3 + g) * H_;
#pragma unroll
        for (int kf = 0; kf < 2; ++kf) {
            const int kb = kf * 32 + hi * 8;
#pragma unroll
            for (int e = 0; e < 16; ++e)
                Bf[s][kf][e] = (__bf16)wrow[kb + e + (e >= 8 ? 8 : 0)];
        }
        const float wih = Wih [c * G3 + g];
        const float bi  = b_ih[c * G3 + g];
        const float bh  = b_hh[c * G3 + g];
        // r/z tiles fold gx + both biases into C-init; n tiles fold only b_hh
        cmul[s]  = (g < 2 * H_) ? wih     : 0.0f;
        cbase[s] = (g < 2 * H_) ? bi + bh : bh;
    }

    // gate-phase per-thread constants (thread owns row m, 8 hidden cols)
    const int m  = tid >> 3;
    const int i0 = (tid & 7) * 8;
    const int myLen = lens[b0 + m];
    float wn[8], bn[8];
#pragma unroll
    for (int i = 0; i < 8; ++i) {
        wn[i] = Wih [c * G3 + 2 * H_ + i0 + i];
        bn[i] = b_ih[c * G3 + 2 * H_ + i0 + i];
    }

    // zero-init hidden state
    for (int k = tid; k < BM * H_; k += 128) {
        hf[k] = 0.0f;
        hb[(k >> 6) * HBW + (k & 63)] = (__bf16)0.0f;
    }
    __syncthreads();

    // ---- time scan --------------------------------------------------------
    for (int t = 0; t < T_; ++t) {
        if (tid < BM)
            xs[tid] = x[((size_t)(b0 + tid) * T_ + t) * C_ + c];
        __syncthreads();                 // prev gate-phase h writes + xs ready
        const float xm = xs[m];          // capture before gate phase uses it

        // A fragments: two 16x32 bf16 chunks of h; elems 0-7 = K kb..kb+7,
        // elems 8-15 = K kb+16..kb+23, kb = 8*hi  (ISA 16-bit A layout)
        const __bf16* hrow = hb + (size_t)col * HBW;
        const int kb = hi * 8;
        const v8bf l0  = *(const v8bf*)(hrow + kb);
        const v8bf h0v = *(const v8bf*)(hrow + kb + 16);
        const v8bf l1  = *(const v8bf*)(hrow + 32 + kb);
        const v8bf h1v = *(const v8bf*)(hrow + 32 + kb + 16);
        const v16bf A0 = __builtin_shufflevector(l0, h0v, 0,1,2,3,4,5,6,7,8,9,10,11,12,13,14,15);
        const v16bf A1 = __builtin_shufflevector(l1, h1v, 0,1,2,3,4,5,6,7,8,9,10,11,12,13,14,15);

#pragma unroll
        for (int s = 0; s < 3; ++s) {
            v8f acc;
#pragma unroll
            for (int j = 0; j < 8; ++j)                       // row m = j + 8*hi
                acc[j] = fmaf(xs[j + hi * 8], cmul[s], cbase[s]);
            acc = __builtin_amdgcn_wmma_f32_16x16x32_bf16(false, A0, false, Bf[s][0],
                                                          (short)0, acc, false, false);
            acc = __builtin_amdgcn_wmma_f32_16x16x32_bf16(false, A1, false, Bf[s][1],
                                                          (short)0, acc, false, false);
            const int g0 = (w * 3 + s) * 16 + col;
#pragma unroll
            for (int j = 0; j < 8; ++j)
                accS[(j + hi * 8) * ACCW + g0] = acc[j];
        }
        __syncthreads();                 // accS ready for gate phase

        // ---- gate phase: all 128 threads, 8 elements each ----
        const bool keep = (t < myLen);
#pragma unroll
        for (int i = 0; i < 8; ++i) {
            const int hc = i0 + i;
            const float ar = accS[m * ACCW + hc];
            const float az = accS[m * ACCW + H_ + hc];
            const float an = accS[m * ACCW + 2 * H_ + hc];   // gh_n (incl. b_hh)
            const float r  = 1.0f / (1.0f + __expf(-ar));
            const float z  = 1.0f / (1.0f + __expf(-az));
            const float pre = fmaf(r, an, fmaf(xm, wn[i], bn[i]));
            const float e2  = __expf(-2.0f * pre);
            const float n   = (1.0f - e2) / (1.0f + e2);     // tanh(pre)
            const float ho  = hf[m * H_ + hc];
            float hv = fmaf(z, ho - n, n);                   // (1-z)n + z h
            hv = keep ? hv : ho;
            hf[m * H_ + hc]  = hv;
            hb[m * HBW + hc] = (__bf16)hv;
        }
        // next iteration's first barrier orders hb/hf writes vs. WMMA reads
    }
    __syncthreads();

    // final hidden state -> hfinal[b][c][h]
#pragma unroll
    for (int i = 0; i < 8; ++i) {
        const int hc = i0 + i;
        hfinal[((size_t)(b0 + m) * C_ + c) * H_ + hc] = hf[m * H_ + hc];
    }
}

// ---------------------------------------------------------------------------
// Kernel 2: proj[b, j] = bp[j] + sum_k hfinal[b,k] * Wp[j,k]   (K = C*H = 2176)
// ---------------------------------------------------------------------------
__global__ __launch_bounds__(64)
void proj_kernel(const float* __restrict__ hfinal,
                 const float* __restrict__ Wp,
                 const float* __restrict__ bp,
                 float*       __restrict__ proj)
{
    const int b = blockIdx.x;
    const int j = threadIdx.x;
    const float* feat = hfinal + (size_t)b * (C_ * H_);
    const float* wrow = Wp     + (size_t)j * (C_ * H_);
    float acc = bp[j];
#pragma unroll 4
    for (int k = 0; k < C_ * H_; ++k)
        acc = fmaf(feat[k], wrow[k], acc);
    proj[b * H_ + j] = acc;
}

// ---------------------------------------------------------------------------
// Kernel 3: batch-norm (batch stats) + ReLU + 1-unit head + sigmoid
// ---------------------------------------------------------------------------
__global__ __launch_bounds__(256)
void head_kernel(const float* __restrict__ proj,
                 const float* __restrict__ gamma,
                 const float* __restrict__ beta,
                 const float* __restrict__ Wpred,
                 const float* __restrict__ bpred,
                 float*       __restrict__ out)
{
    __shared__ float muS[H_], rsS[H_], gS[H_], btS[H_], wpS[H_];
    const int tid = threadIdx.x;
    if (tid < H_) {
        float s = 0.0f, s2 = 0.0f;
        for (int b = 0; b < B_; ++b) {
            const float v = proj[b * H_ + tid];
            s += v;
            s2 = fmaf(v, v, s2);
        }
        const float mu  = s * (1.0f / B_);
        const float var = s2 * (1.0f / B_) - mu * mu;        // biased, as torch BN
        muS[tid] = mu;
        rsS[tid] = rsqrtf(var + 1e-5f);
        gS[tid]  = gamma[tid];
        btS[tid] = beta[tid];
        wpS[tid] = Wpred[tid];
    }
    __syncthreads();
    float acc = bpred[0];
#pragma unroll 4
    for (int j = 0; j < H_; ++j) {
        float y = fmaf(gS[j] * (proj[tid * H_ + j] - muS[j]), rsS[j], btS[j]);
        y = fmaxf(y, 0.0f);
        acc = fmaf(y, wpS[j], acc);
    }
    out[tid] = 1.0f / (1.0f + __expf(-acc));
}

// ---------------------------------------------------------------------------
extern "C" void kernel_launch(void* const* d_in, const int* in_sizes, int n_in,
                              void* d_out, int out_size, void* d_ws, size_t ws_size,
                              hipStream_t stream)
{
    const float* x     = (const float*)d_in[0];
    const float* Wih   = (const float*)d_in[1];
    const float* Whh   = (const float*)d_in[2];
    const float* b_ih  = (const float*)d_in[3];
    const float* b_hh  = (const float*)d_in[4];
    const float* Wp    = (const float*)d_in[5];
    const float* bp    = (const float*)d_in[6];
    const float* gamma = (const float*)d_in[7];
    const float* beta  = (const float*)d_in[8];
    const float* Wpred = (const float*)d_in[9];
    const float* bpred = (const float*)d_in[10];
    const int*   lens  = (const int*)d_in[11];

    float* hfinal = (float*)d_ws;                        // B*C*H floats (~2.2 MB)
    float* proj   = hfinal + (size_t)B_ * C_ * H_;       // B*H floats
    float* out    = (float*)d_out;                       // [B] fp32

    gru_scan_kernel<<<C_ * (B_ / BM), 128, 0, stream>>>(x, Wih, Whh, b_ih, b_hh, lens, hfinal);
    proj_kernel   <<<B_, 64, 0, stream>>>(hfinal, Wp, bp, proj);
    head_kernel   <<<1, B_, 0, stream>>>(proj, gamma, beta, Wpred, bpred, out);
}
